// NerfFusionHead_14766097563777
// MI455X (gfx1250) — compile-verified, matrix operand order checked by reading, not run on hardware
//
#include <hip/hip_runtime.h>
#include <hip/hip_bf16.h>

// ---------------------------------------------------------------------------
// Types
// ---------------------------------------------------------------------------
typedef __attribute__((ext_vector_type(16))) _Float16     v16h;
typedef __attribute__((ext_vector_type(8)))  float        v8f;
typedef __attribute__((ext_vector_type(4)))  unsigned int v4u;
typedef __attribute__((ext_vector_type(4)))  float        v4f;

union V16H { v16h v; unsigned int u[8]; v4u q[2]; };

// ---------------------------------------------------------------------------
// Problem dims
// ---------------------------------------------------------------------------
#define RRAYS 3600
#define PSAMP 64
#define NPTS  (RRAYS * PSAMP)      // 230400
#define DLAT  768
#define DHID  256
#define MTILE 64                   // points per workgroup
#define KIN_PAD 96                 // 78 / 84 padded to 96 (3 k-tiles of 32)

// packed-weight dword offsets within one MLP region (dwords = half2 pairs)
#define PW_WIN    0
#define PW_WIN_SZ 12288            // 16ct * 3kt * 32lanes * 8
#define PW_WZ     12288
#define PW_WZ_SZ  98304            // 16 * 24 * 32 * 8
#define PW_W0     (12288 + 3 * 98304)          // 307200
#define PW_W0_SZ  32768            // 16 * 8 * 32 * 8
#define PW_W1     (307200 + 3 * 32768)         // 405504
#define PW_MLP_DWORDS (405504 + 3 * 32768)     // 503808 dwords = 2,015,232 B

// ws byte offsets
#define WS_PWD    0u
#define WS_PWC    ((size_t)PW_MLP_DWORDS * 4)  // 2,015,232
#define WS_SIGMA  (size_t)4194304
#define WS_RGB    (size_t)5242880

#define SMEM_BYTES ((MTILE * DLAT + MTILE * DHID) * 2)   // 131072

// ---------------------------------------------------------------------------
// Weight pack: fp32 (K x 256) row-major  ->  f16 B-fragment layout
// frag dword index: ((ct*Ktiles + kt)*32 + lane)*8 + j
// lane: col = ct*16 + (lane&15); kbase = kt*32 + (lane>>4)*8
// j<4 : K = kbase + 2j,2j+1 ; j>=4 : K = kbase + 16 + 2(j-4), +1
// ---------------------------------------------------------------------------
__device__ __forceinline__ unsigned short h2u(_Float16 h) {
  union { _Float16 h; unsigned short u; } c; c.h = h; return c.u;
}

__global__ void pack_w_kernel(const float* __restrict__ W, unsigned int* __restrict__ dst,
                              int Kact, int Ktiles) {
  int id = blockIdx.x * blockDim.x + threadIdx.x;
  int total = 16 * Ktiles * 32;
  if (id >= total) return;
  int lane = id & 31;
  int kt   = (id >> 5) % Ktiles;
  int ct   = (id >> 5) / Ktiles;
  int kb   = kt * 32 + ((lane >> 4) << 3);
  int n    = ct * 16 + (lane & 15);
  unsigned int* out = dst + (size_t)(((ct * Ktiles) + kt) * 32 + lane) * 8;
#pragma unroll
  for (int j = 0; j < 8; ++j) {
    int k = kb + ((j < 4) ? (2 * j) : (16 + 2 * (j - 4)));
    float a = (k     < Kact) ? W[(size_t)k       * DHID + n] : 0.f;
    float b = (k + 1 < Kact) ? W[(size_t)(k + 1) * DHID + n] : 0.f;
    unsigned int lo = h2u((_Float16)a);
    unsigned int hi = h2u((_Float16)b);
    out[j] = lo | (hi << 16);
  }
}

// ---------------------------------------------------------------------------
// WMMA fragment helpers (wave32, V_WMMA_F32_16X16X32_F16)
// ---------------------------------------------------------------------------
__device__ __forceinline__ v16h load_a_frag(const _Float16* buf, int Kpad,
                                            int rowbase, int kt, int lane) {
  int row = rowbase + (lane & 15);
  const _Float16* p = buf + row * Kpad + kt * 32 + ((lane >> 4) << 3);
  V16H r;
#pragma unroll
  for (int j = 0; j < 4; ++j) {
    r.u[j]     = *(const unsigned int*)(p + 2 * j);        // K = kb+2j, kb+2j+1
    r.u[j + 4] = *(const unsigned int*)(p + 16 + 2 * j);   // K = kb+16+2j, +1
  }
  return r.v;
}

__device__ __forceinline__ v16h load_b_frag(const unsigned int* pk, int Ktiles,
                                            int ct, int kt, int lane) {
  const v4u* p = (const v4u*)(pk + (size_t)(((ct * Ktiles) + kt) * 32 + lane) * 8);
  V16H r;
  r.q[0] = p[0];
  r.q[1] = p[1];
  return r.v;
}

__device__ __forceinline__ void gemm_tiles(v8f* acc, const _Float16* A, int Kpad,
                                           int rowbase, const unsigned int* Bp,
                                           int Ktiles, int ct_base, int lane) {
#pragma unroll 1
  for (int kt = 0; kt < Ktiles; ++kt) {
    v16h a = load_a_frag(A, Kpad, rowbase, kt, lane);
#pragma unroll
    for (int cc = 0; cc < 8; ++cc) {
      v16h b = load_b_frag(Bp, Ktiles, ct_base + cc, kt, lane);
      acc[cc] = __builtin_amdgcn_wmma_f32_16x16x32_f16(
          false, a, false, b, (short)0, acc[cc], false, false);
    }
  }
}

// C/D layout: lane l, vgpr v -> row = v + 8*(l>>4), col = (l&15) (+16 per tile)
__device__ __forceinline__ void set_bias(v8f* acc, const float* bias, int col0, int lane) {
  int c = lane & 15;
#pragma unroll
  for (int cc = 0; cc < 8; ++cc) {
    float bv = bias[col0 + cc * 16 + c];
#pragma unroll
    for (int v = 0; v < 8; ++v) acc[cc][v] = bv;
  }
}

__device__ __forceinline__ void add_bias(v8f* acc, const float* bias, int col0, int lane) {
  int c = lane & 15;
#pragma unroll
  for (int cc = 0; cc < 8; ++cc) {
    float bv = bias[col0 + cc * 16 + c];
#pragma unroll
    for (int v = 0; v < 8; ++v) acc[cc][v] += bv;
  }
}

__device__ __forceinline__ void store_relu(const v8f* acc, _Float16* act,
                                           int rowbase, int col0, int lane) {
  int r0 = rowbase + ((lane >> 4) << 3);
  int c0 = col0 + (lane & 15);
#pragma unroll
  for (int cc = 0; cc < 8; ++cc) {
#pragma unroll
    for (int v = 0; v < 8; ++v) {
      float x = acc[cc][v];
      x = x > 0.f ? x : 0.f;
      act[(size_t)(r0 + v) * DHID + c0 + cc * 16] = (_Float16)x;
    }
  }
}

// ---------------------------------------------------------------------------
// ResnetFC core (both MLPs identical shape after K padding)
// ---------------------------------------------------------------------------
__device__ __forceinline__ void run_mlp(const _Float16* z, _Float16* act,
                                        const unsigned int* pw,
                                        const float* b_in, const float* bz,
                                        const float* b0, const float* b1,
                                        int rowbase, int ct_base, int col0, int lane,
                                        v8f* xacc) {
  // x = X_in @ W_in + b_in  (A = xin in act region, stride KIN_PAD)
  set_bias(xacc, b_in, col0, lane);
  gemm_tiles(xacc, act, KIN_PAD, rowbase, pw + PW_WIN, 3, ct_base, lane);

#pragma unroll 1
  for (int i = 0; i < 3; ++i) {
    // x += z @ Wz[i] + bz[i]
    add_bias(xacc, bz + i * DHID, col0, lane);
    gemm_tiles(xacc, z, DLAT, rowbase, pw + PW_WZ + i * PW_WZ_SZ, 24, ct_base, lane);

    // net = relu(x) @ W0[i] + b0[i]
    v8f nacc[8];
    set_bias(nacc, b0 + i * DHID, col0, lane);
    __syncthreads();
    store_relu(xacc, act, rowbase, col0, lane);
    __syncthreads();
    gemm_tiles(nacc, act, DHID, rowbase, pw + PW_W0 + i * PW_W0_SZ, 8, ct_base, lane);

    // x += relu(net) @ W1[i] + b1[i]
    add_bias(xacc, b1 + i * DHID, col0, lane);
    __syncthreads();
    store_relu(nacc, act, rowbase, col0, lane);
    __syncthreads();
    gemm_tiles(xacc, act, DHID, rowbase, pw + PW_W1 + i * PW_W0_SZ, 8, ct_base, lane);
  }

  // leave relu(x) in act for the head
  __syncthreads();
  store_relu(xacc, act, rowbase, col0, lane);
  __syncthreads();
}

// ---------------------------------------------------------------------------
// Positional encoding: [x,y,z, (sin sxyz, cos cxyz) x 6 freqs] = 39 halves
// ---------------------------------------------------------------------------
__device__ __forceinline__ void posenc39(const float* __restrict__ p,
                                         _Float16* __restrict__ dst) {
  float x = p[0], y = p[1], z = p[2];
  dst[0] = (_Float16)x; dst[1] = (_Float16)y; dst[2] = (_Float16)z;
  float f = 3.14159265358979f;
  int o = 3;
#pragma unroll
  for (int k = 0; k < 6; ++k) {
    dst[o + 0] = (_Float16)__sinf(f * x);
    dst[o + 1] = (_Float16)__sinf(f * y);
    dst[o + 2] = (_Float16)__sinf(f * z);
    dst[o + 3] = (_Float16)__cosf(f * x);
    dst[o + 4] = (_Float16)__cosf(f * y);
    dst[o + 5] = (_Float16)__cosf(f * z);
    o += 6;
    f *= 2.f;
  }
}

// ---------------------------------------------------------------------------
// Fused MLP kernel: 1 workgroup = 64 points, 8 waves (4 row-tiles x 2 col-halves)
// ---------------------------------------------------------------------------
__global__ void __launch_bounds__(256) nerf_mlp_kernel(
    const float* __restrict__ pts_a, const float* __restrict__ pts_b,
    const float* __restrict__ dirs_a, const float* __restrict__ dirs_b,
    const float* __restrict__ latent,
    const unsigned int* __restrict__ pw_d, const unsigned int* __restrict__ pw_c,
    const float* __restrict__ db_in, const float* __restrict__ dbz,
    const float* __restrict__ db0,  const float* __restrict__ db1,
    const float* __restrict__ cb_in, const float* __restrict__ cbz,
    const float* __restrict__ cb0,  const float* __restrict__ cb1,
    const float* __restrict__ dW_out, const float* __restrict__ db_out,
    const float* __restrict__ cW_out, const float* __restrict__ cb_out,
    float* __restrict__ sigma_out, float* __restrict__ rgb_out) {
  extern __shared__ char smem[];
  _Float16* zsh = (_Float16*)smem;                           // [64][768] f16
  _Float16* act = (_Float16*)(smem + MTILE * DLAT * 2);      // [64][256] f16 (aliases xin [64][96])

  const int tid  = threadIdx.x;
  const int lane = tid & 31;
  const int wv   = tid >> 5;
  const int rowbase = (wv & 3) * 16;
  const int ch      = wv >> 2;
  const int ct_base = ch * 8;
  const int col0    = ch * 128;
  const size_t pbase = (size_t)blockIdx.x * MTILE;

  // ---- latent fp32 -> LDS f16 (read once from HBM) ----
  {
    const v4f* src = (const v4f*)(latent + pbase * DLAT);
#pragma unroll 1
    for (int q = tid; q < MTILE * DLAT / 4; q += 256) {
      v4f v = src[q];
      _Float16* d = zsh + (size_t)q * 4;
      d[0] = (_Float16)v[0]; d[1] = (_Float16)v[1];
      d[2] = (_Float16)v[2]; d[3] = (_Float16)v[3];
    }
  }
  // ---- density input: [ea(39) | eb(39) | pad -> 96] ----
  {
    int p = tid >> 2, seg = tid & 3;
    size_t gp = pbase + p;
    _Float16* row = act + p * KIN_PAD;
    if (seg == 0) posenc39(pts_a + gp * 3, row);
    else if (seg == 1) posenc39(pts_b + gp * 3, row + 39);
    else if (seg == 2) { for (int c = 78; c < KIN_PAD; ++c) row[c] = (_Float16)0.f; }
  }
  __syncthreads();

  v8f xacc[8];

  // ===== density MLP =====
  run_mlp(zsh, act, pw_d, db_in, dbz, db0, db1, rowbase, ct_base, col0, lane, xacc);
  if (tid < MTILE) {
    const _Float16* a = act + (size_t)tid * DHID;
    float s = db_out[0];
#pragma unroll 4
    for (int c = 0; c < DHID; ++c) s += (float)a[c] * dW_out[c];
    sigma_out[pbase + tid] = s;
  }
  __syncthreads();

  // ---- color input: [ea(39) | dirs_a(3) | eb(39) | dirs_b(3) | pad -> 96] ----
  {
    int p = tid >> 2, seg = tid & 3;
    size_t gp = pbase + p;
    _Float16* row = act + p * KIN_PAD;
    if (seg == 0) posenc39(pts_a + gp * 3, row);
    else if (seg == 1) {
      row[39] = (_Float16)dirs_a[gp * 3 + 0];
      row[40] = (_Float16)dirs_a[gp * 3 + 1];
      row[41] = (_Float16)dirs_a[gp * 3 + 2];
      for (int c = 84; c < KIN_PAD; ++c) row[c] = (_Float16)0.f;
    } else if (seg == 2) {
      posenc39(pts_b + gp * 3, row + 42);
    } else {
      row[81] = (_Float16)dirs_b[gp * 3 + 0];
      row[82] = (_Float16)dirs_b[gp * 3 + 1];
      row[83] = (_Float16)dirs_b[gp * 3 + 2];
    }
  }
  __syncthreads();

  // ===== color MLP =====
  run_mlp(zsh, act, pw_c, cb_in, cbz, cb0, cb1, rowbase, ct_base, col0, lane, xacc);
  if (tid < MTILE) {
    const _Float16* a = act + (size_t)tid * DHID;
    float s0 = cb_out[0], s1 = cb_out[1], s2 = cb_out[2];
#pragma unroll 4
    for (int c = 0; c < DHID; ++c) {
      float av = (float)a[c];
      s0 += av * cW_out[c * 3 + 0];
      s1 += av * cW_out[c * 3 + 1];
      s2 += av * cW_out[c * 3 + 2];
    }
    float* o = rgb_out + (pbase + tid) * 3;
    o[0] = 1.f / (1.f + __expf(-s0));
    o[1] = 1.f / (1.f + __expf(-s1));
    o[2] = 1.f / (1.f + __expf(-s2));
  }
}

// ---------------------------------------------------------------------------
// Volume rendering: one thread per ray, sequential scan over P=64
// ---------------------------------------------------------------------------
__global__ void render_kernel(const float* __restrict__ z_vals,
                              const float* __restrict__ sigma,
                              const float* __restrict__ rgb,
                              float* __restrict__ out) {
  int r = blockIdx.x * blockDim.x + threadIdx.x;
  if (r >= RRAYS) return;
  const float* zr = z_vals + (size_t)r * PSAMP;
  const float* sr = sigma + (size_t)r * PSAMP;
  const float* cr = rgb + (size_t)r * PSAMP * 3;
  float T = 1.f, c0 = 0.f, c1 = 0.f, c2 = 0.f, dep = 0.f;
  float zc = zr[0];
#pragma unroll 1
  for (int p = 0; p < PSAMP; ++p) {
    float zn = (p < PSAMP - 1) ? zr[p + 1] : 0.f;
    float delta = (p < PSAMP - 1) ? (zn - zc) : 1e3f;
    float s = sr[p];
    s = s > 0.f ? s : 0.f;
    float alpha = 1.f - __expf(-s * delta);
    float w = alpha * T;
    T *= (1.f - alpha + 1e-10f);
    c0 += w * cr[p * 3 + 0];
    c1 += w * cr[p * 3 + 1];
    c2 += w * cr[p * 3 + 2];
    dep += w * zc;
    zc = zn;
  }
  float* o = out + (size_t)r * 4;
  o[0] = c0; o[1] = c1; o[2] = c2; o[3] = dep;
}

// ---------------------------------------------------------------------------
// Host entry
// ---------------------------------------------------------------------------
extern "C" void kernel_launch(void* const* d_in, const int* in_sizes, int n_in,
                              void* d_out, int out_size, void* d_ws, size_t ws_size,
                              hipStream_t stream) {
  (void)in_sizes; (void)n_in; (void)out_size; (void)ws_size;
  const float* pts_a  = (const float*)d_in[0];
  const float* pts_b  = (const float*)d_in[1];
  const float* dirs_a = (const float*)d_in[2];
  const float* dirs_b = (const float*)d_in[3];
  const float* latent = (const float*)d_in[4];
  const float* z_vals = (const float*)d_in[5];
  const float* dW_in  = (const float*)d_in[6];
  const float* db_in  = (const float*)d_in[7];
  const float* dWz    = (const float*)d_in[8];
  const float* dbz    = (const float*)d_in[9];
  const float* dW0    = (const float*)d_in[10];
  const float* db0    = (const float*)d_in[11];
  const float* dW1    = (const float*)d_in[12];
  const float* db1    = (const float*)d_in[13];
  const float* cW_in  = (const float*)d_in[14];
  const float* cb_in  = (const float*)d_in[15];
  const float* cWz    = (const float*)d_in[16];
  const float* cbz    = (const float*)d_in[17];
  const float* cW0    = (const float*)d_in[18];
  const float* cb0    = (const float*)d_in[19];
  const float* cW1    = (const float*)d_in[20];
  const float* cb1    = (const float*)d_in[21];
  const float* dW_out = (const float*)d_in[22];
  const float* db_out = (const float*)d_in[23];
  const float* cW_out = (const float*)d_in[24];
  const float* cb_out = (const float*)d_in[25];

  char* ws = (char*)d_ws;
  unsigned int* pw_d = (unsigned int*)(ws + WS_PWD);
  unsigned int* pw_c = (unsigned int*)(ws + WS_PWC);
  float* sigma = (float*)(ws + WS_SIGMA);
  float* rgb   = (float*)(ws + WS_RGB);

  auto pack = [&](const float* W, unsigned int* dst, int Kact, int Ktiles) {
    int total = 16 * Ktiles * 32;
    pack_w_kernel<<<(total + 255) / 256, 256, 0, stream>>>(W, dst, Kact, Ktiles);
  };
  pack(dW_in, pw_d + PW_WIN, 78, 3);
  pack(cW_in, pw_c + PW_WIN, 84, 3);
  for (int i = 0; i < 3; ++i) {
    pack(dWz + (size_t)i * DLAT * DHID, pw_d + PW_WZ + i * PW_WZ_SZ, DLAT, 24);
    pack(cWz + (size_t)i * DLAT * DHID, pw_c + PW_WZ + i * PW_WZ_SZ, DLAT, 24);
    pack(dW0 + (size_t)i * DHID * DHID, pw_d + PW_W0 + i * PW_W0_SZ, DHID, 8);
    pack(cW0 + (size_t)i * DHID * DHID, pw_c + PW_W0 + i * PW_W0_SZ, DHID, 8);
    pack(dW1 + (size_t)i * DHID * DHID, pw_d + PW_W1 + i * PW_W0_SZ, DHID, 8);
    pack(cW1 + (size_t)i * DHID * DHID, pw_c + PW_W1 + i * PW_W0_SZ, DHID, 8);
  }

  hipFuncSetAttribute((const void*)nerf_mlp_kernel,
                      hipFuncAttributeMaxDynamicSharedMemorySize, SMEM_BYTES);
  nerf_mlp_kernel<<<NPTS / MTILE, 256, SMEM_BYTES, stream>>>(
      pts_a, pts_b, dirs_a, dirs_b, latent, pw_d, pw_c,
      db_in, dbz, db0, db1, cb_in, cbz, cb0, cb1,
      dW_out, db_out, cW_out, cb_out, sigma, rgb);

  render_kernel<<<(RRAYS + 255) / 256, 256, 0, stream>>>(z_vals, sigma, rgb, (float*)d_out);
}
